// MyLSTM_10067403341984
// MI455X (gfx1250) — matrix operand order repaired and checked by reading, not compiled, not run on hardware
//
#include <hip/hip_runtime.h>
#include <hip/hip_bf16.h>

typedef __attribute__((ext_vector_type(16))) __bf16 v16bf;
typedef __attribute__((ext_vector_type(8)))  __bf16 v8bf;
typedef __attribute__((ext_vector_type(8)))  float  v8f;

// GCC-vector int4 to match the async-builtin's parameter type exactly
typedef int v4i_ __attribute__((vector_size(16)));
typedef __attribute__((address_space(1))) v4i_* gas_v4p;
typedef __attribute__((address_space(3))) v4i_* las_v4p;

#define LSTM_B   64
#define LSTM_T   512
#define LSTM_D   512
#define LSTM_H   1024
#define LSTM_G   4096
#define NWG      64
#define TPB      128

__device__ __forceinline__ float sigf(float x)      { return 1.0f / (1.0f + __expf(-x)); }
__device__ __forceinline__ float tanh_fast(float x) { return 1.0f - 2.0f / (__expf(2.0f * x) + 1.0f); }

// ---------------- fp32 -> bf16 conversion ----------------
__global__ void cvt_bf16_kernel(const float* __restrict__ s, __bf16* __restrict__ d, int n) {
  for (int i = blockIdx.x * blockDim.x + threadIdx.x; i < n; i += gridDim.x * blockDim.x)
    d[i] = (__bf16)s[i];
}

// ---------------- bias sum + state zero-init + barrier reset ----------------
__global__ void prep_kernel(const float* __restrict__ bih, const float* __restrict__ bhh,
                            float* __restrict__ bias, __bf16* __restrict__ hb0,
                            __bf16* __restrict__ hb1, float* __restrict__ cbuf,
                            unsigned* __restrict__ cnt) {
  int i = blockIdx.x * blockDim.x + threadIdx.x;
  if (i < LSTM_G) bias[i] = bih[i] + bhh[i];
  if (i < LSTM_B * LSTM_H) {
    hb0[i] = (__bf16)0.0f;
    hb1[i] = (__bf16)0.0f;
    cbuf[i] = 0.0f;
  }
  if (i == 0) *cnt = 0u;
}

// ---------------- persistent fused LSTM scan ----------------
// Grid: 64 WGs, WG wg owns hidden columns [wg*16, wg*16+16).
// Each step: g = [x_t | h] @ [W_ih; W_hh]^T + bias for its 4 gate tiles (i,f,g,o),
// gating in registers, write h/c/output, one grid barrier.
__global__ void __launch_bounds__(TPB, 1)
lstm_scan_kernel(const __bf16* __restrict__ xb, const __bf16* __restrict__ Wihb,
                 const __bf16* __restrict__ Whhb, const float* __restrict__ bias,
                 __bf16* __restrict__ hb0, __bf16* __restrict__ hb1,
                 float* __restrict__ cbuf, float* __restrict__ out,
                 float* __restrict__ outh, float* __restrict__ outc,
                 unsigned* __restrict__ cnt)
{
  __shared__ uint4 smem4[8192];           // 128 KB: W_hh slice, 64 rows x 1024 bf16, xor-swizzled
  char* smem = (char*)smem4;

  const int j0   = blockIdx.x * 16;       // owned hidden-column base
  const int tid  = threadIdx.x;
  const int wv   = tid >> 5;              // wave id 0..3  -> batch block
  const int ln   = tid & 31;              // lane in wave
  const int r    = ln & 15;               // row/col within 16
  const int half = ln >> 4;               // lane half
  const int m0   = wv * 16;               // batch tile base
  const int sw   = (r & 7) << 5;          // LDS xor swizzle for this lane's row

  // Stage W_hh rows {g*1024 + j0 + 0..15 | g=0..3} into LDS (16B granules, swizzled).
  // CDNA5 async LDS-load path (ASYNCcnt-tracked, no VGPR round trip); per-lane LDS
  // destination addresses let us keep the bank swizzle.
#if __has_builtin(__builtin_amdgcn_global_load_async_to_lds_b128)
  for (int c2 = tid; c2 < 64 * 128; c2 += TPB) {
    int row  = c2 >> 7;                   // 0..63 local row
    int kb   = (c2 & 127) << 4;           // byte offset in row, 16B granule
    int grow = ((row >> 4) * LSTM_H) + j0 + (row & 15);
    const char* gsrc = (const char*)Whhb + (size_t)grow * (LSTM_H * 2) + kb;
    char*       ldst = smem + row * 2048 + (kb ^ ((row & 7) << 5));
    gas_v4p g = (gas_v4p)(unsigned long long)(uintptr_t)gsrc;
    las_v4p l = (las_v4p)(unsigned)(uintptr_t)ldst;   // low 32 bits of generic LDS ptr == LDS offset
    __builtin_amdgcn_global_load_async_to_lds_b128(g, l, 0, 0);
  }
#if __has_builtin(__builtin_amdgcn_s_wait_asynccnt)
  __builtin_amdgcn_s_wait_asynccnt(0);
#else
  asm volatile("s_wait_asynccnt 0" ::: "memory");
#endif
#else
  for (int c2 = tid; c2 < 64 * 128; c2 += TPB) {
    int row  = c2 >> 7;
    int kb   = (c2 & 127) << 4;
    int grow = ((row >> 4) * LSTM_H) + j0 + (row & 15);
    uint4 v = *(const uint4*)((const char*)Whhb + (size_t)grow * (LSTM_H * 2) + kb);
    *(uint4*)(smem + row * 2048 + (kb ^ ((row & 7) << 5))) = v;
  }
#endif
  __syncthreads();

  // Bias per gate for this lane's output column n = gate*H + j0 + (lane&15)
  const float bi  = bias[0 * LSTM_H + j0 + r];
  const float bff = bias[1 * LSTM_H + j0 + r];
  const float bg  = bias[2 * LSTM_H + j0 + r];
  const float bo  = bias[3 * LSTM_H + j0 + r];

  for (int t = 0; t < LSTM_T; ++t) {
    const __bf16* hread  = (t & 1) ? hb1 : hb0;
    __bf16*       hwrite = (t & 1) ? hb0 : hb1;

    v8f a0 = {0,0,0,0,0,0,0,0};
    v8f a1 = a0, a2 = a0, a3 = a0;

    // ---- K over D (x_t part), weights from global (L2-resident) W_ih ----
    const __bf16* xrow = xb + (size_t)(m0 + r) * (LSTM_T * LSTM_D) + (size_t)t * LSTM_D;
    // Hide next step's HBM first-touch of x behind this step's GEMM.
    if (t + 1 < LSTM_T) {
      __builtin_prefetch(xrow + LSTM_D, 0, 0);
      __builtin_prefetch(xrow + LSTM_D + 256, 0, 0);
    }
    const size_t wr = (size_t)(j0 + r) * LSTM_D;
    #pragma unroll 4
    for (int k0 = 0; k0 < LSTM_D; k0 += 32) {
      v8bf alo = *(const v8bf*)(xrow + k0 + half * 16);
      v8bf ahi = *(const v8bf*)(xrow + k0 + half * 16 + 8);
      v16bf a = __builtin_shufflevector(alo, ahi, 0,1,2,3,4,5,6,7,8,9,10,11,12,13,14,15);
      const __bf16* wp = Wihb + wr + k0 + half * 16;
      v16bf b0 = *(const v16bf*)(wp);
      v16bf b1 = *(const v16bf*)(wp + (size_t)1 * LSTM_H * LSTM_D);
      v16bf b2 = *(const v16bf*)(wp + (size_t)2 * LSTM_H * LSTM_D);
      v16bf b3 = *(const v16bf*)(wp + (size_t)3 * LSTM_H * LSTM_D);
      a0 = __builtin_amdgcn_wmma_f32_16x16x32_bf16(false, a, false, b0, (short)0, a0, false, false);
      a1 = __builtin_amdgcn_wmma_f32_16x16x32_bf16(false, a, false, b1, (short)0, a1, false, false);
      a2 = __builtin_amdgcn_wmma_f32_16x16x32_bf16(false, a, false, b2, (short)0, a2, false, false);
      a3 = __builtin_amdgcn_wmma_f32_16x16x32_bf16(false, a, false, b3, (short)0, a3, false, false);
    }

    // ---- K over H (h part), weights from LDS ----
    const __bf16* hrow = hread + (size_t)(m0 + r) * LSTM_H;
    #pragma unroll 4
    for (int k0 = 0; k0 < LSTM_H; k0 += 32) {
      v8bf alo = *(const v8bf*)(hrow + k0 + half * 16);
      v8bf ahi = *(const v8bf*)(hrow + k0 + half * 16 + 8);
      v16bf a = __builtin_shufflevector(alo, ahi, 0,1,2,3,4,5,6,7,8,9,10,11,12,13,14,15);
      const int kb = ((k0 + half * 16) * 2) ^ sw;   // 32B aligned, swizzle-consistent
      v16bf b0 = *(const v16bf*)(smem + (0 * 16 + r) * 2048 + kb);
      v16bf b1 = *(const v16bf*)(smem + (1 * 16 + r) * 2048 + kb);
      v16bf b2 = *(const v16bf*)(smem + (2 * 16 + r) * 2048 + kb);
      v16bf b3 = *(const v16bf*)(smem + (3 * 16 + r) * 2048 + kb);
      a0 = __builtin_amdgcn_wmma_f32_16x16x32_bf16(false, a, false, b0, (short)0, a0, false, false);
      a1 = __builtin_amdgcn_wmma_f32_16x16x32_bf16(false, a, false, b1, (short)0, a1, false, false);
      a2 = __builtin_amdgcn_wmma_f32_16x16x32_bf16(false, a, false, b2, (short)0, a2, false, false);
      a3 = __builtin_amdgcn_wmma_f32_16x16x32_bf16(false, a, false, b3, (short)0, a3, false, false);
    }

    // ---- gating directly on accumulator registers (D layout: m=8*half+e, n=j0+r) ----
    #pragma unroll
    for (int e = 0; e < 8; ++e) {
      const int b = m0 + 8 * half + e;
      const int j = j0 + r;
      const size_t cidx = (size_t)b * LSTM_H + j;
      float iv = sigf(a0[e] + bi);
      float fv = sigf(a1[e] + bff);
      float gv = tanh_fast(a2[e] + bg);
      float ov = sigf(a3[e] + bo);
      float cn = fmaf(fv, cbuf[cidx], iv * gv);
      float hn = ov * tanh_fast(cn);
      cbuf[cidx]   = cn;
      hwrite[cidx] = (__bf16)hn;
      out[(size_t)b * ((size_t)LSTM_T * LSTM_H) + (size_t)t * LSTM_H + j] = hn;
      if (t == LSTM_T - 1) { outh[cidx] = hn; outc[cidx] = cn; }
    }

    // ---- grid-wide barrier (monotone counter, release/acquire, agent scope) ----
    __syncthreads();
    if (tid == 0) {
      __hip_atomic_fetch_add(cnt, 1u, __ATOMIC_RELEASE, __HIP_MEMORY_SCOPE_AGENT);
      const unsigned target = (unsigned)(t + 1) * NWG;
      while (__hip_atomic_load(cnt, __ATOMIC_ACQUIRE, __HIP_MEMORY_SCOPE_AGENT) < target)
        __builtin_amdgcn_s_sleep(2);
    }
    __syncthreads();
  }
}

extern "C" void kernel_launch(void* const* d_in, const int* in_sizes, int n_in,
                              void* d_out, int out_size, void* d_ws, size_t ws_size,
                              hipStream_t stream) {
  (void)in_sizes; (void)n_in; (void)out_size; (void)ws_size;
  const float* x    = (const float*)d_in[0];   // [B,T,D]
  const float* W_ih = (const float*)d_in[1];   // [4H,D]
  const float* b_ih = (const float*)d_in[2];   // [4H]
  const float* W_hh = (const float*)d_in[3];   // [4H,H]
  const float* b_hh = (const float*)d_in[4];   // [4H]

  char* ws = (char*)d_ws;                       // ~46.7 MB used
  __bf16*   xb   = (__bf16*)(ws);               // 33,554,432 B
  __bf16*   Wihb = (__bf16*)(ws + 33554432);    //  4,194,304 B
  __bf16*   Whhb = (__bf16*)(ws + 37748736);    //  8,388,608 B
  float*    bias = (float*) (ws + 46137344);    //     16,384 B
  __bf16*   hb0  = (__bf16*)(ws + 46153728);    //    131,072 B
  __bf16*   hb1  = (__bf16*)(ws + 46284800);    //    131,072 B
  float*    cbuf = (float*) (ws + 46415872);    //    262,144 B
  unsigned* cnt  = (unsigned*)(ws + 46678016);  //         64 B

  float* out  = (float*)d_out;                                  // [B,T,H]
  float* outh = out  + (size_t)LSTM_B * LSTM_T * LSTM_H;        // [B,H]
  float* outc = outh + (size_t)LSTM_B * LSTM_H;                 // [B,H]

  cvt_bf16_kernel<<<2048, 256, 0, stream>>>(x,    xb,   LSTM_B * LSTM_T * LSTM_D);
  cvt_bf16_kernel<<<512,  256, 0, stream>>>(W_ih, Wihb, LSTM_G * LSTM_D);
  cvt_bf16_kernel<<<1024, 256, 0, stream>>>(W_hh, Whhb, LSTM_G * LSTM_H);
  prep_kernel<<<256, 256, 0, stream>>>(b_ih, b_hh, bias, hb0, hb1, cbuf, cnt);
  lstm_scan_kernel<<<NWG, TPB, 0, stream>>>(xb, Wihb, Whhb, bias, hb0, hb1, cbuf,
                                            out, outh, outc, cnt);
}
